// CrossHeightAttention_67551245631849
// MI455X (gfx1250) — matrix operand order repaired and checked by reading, not compiled, not run on hardware
//
#include <hip/hip_runtime.h>
#include <hip/hip_bf16.h>

typedef __bf16 bf16_t;
typedef bf16_t v16bf __attribute__((ext_vector_type(16)));
typedef bf16_t v8bf  __attribute__((ext_vector_type(8)));
typedef float  v8f   __attribute__((ext_vector_type(8)));
typedef float  v4f   __attribute__((ext_vector_type(4)));

// ---------------- layout transform kernels ----------------

// out[b][p][c] = in[b][c][p]   (NCHW f32 -> NHWC bf16)
__global__ __launch_bounds__(256)
void nchw_to_nhwc_bf16(const float* __restrict__ in, bf16_t* __restrict__ out,
                       int B, int C, int HW) {
    long i = (long)blockIdx.x * blockDim.x + threadIdx.x;
    long total = (long)B * C * HW;
    if (i >= total) return;
    int c = (int)(i % C);
    long rem = i / C;
    int p = (int)(rem % HW);
    int b = (int)(rem / HW);
    out[i] = (bf16_t)in[((long)b * C + c) * HW + p];
}

// out[b][p][c] = in[b][c][p]   (NCHW f32 -> NHWC f32)
__global__ __launch_bounds__(256)
void nchw_to_nhwc_f32(const float* __restrict__ in, float* __restrict__ out,
                      int B, int C, int HW) {
    long i = (long)blockIdx.x * blockDim.x + threadIdx.x;
    long total = (long)B * C * HW;
    if (i >= total) return;
    int c = (int)(i % C);
    long rem = i / C;
    int p = (int)(rem % HW);
    int b = (int)(rem / HW);
    out[i] = in[((long)b * C + c) * HW + p];
}

// wt[(tap*O + o)*Cin + ci] = w[(o*Cin + ci)*9 + tap]   (OIHW f32 -> [tap][O][Cin] bf16)
__global__ __launch_bounds__(256)
void weight_transform(const float* __restrict__ w, bf16_t* __restrict__ wt,
                      int O, int Cin) {
    long i = (long)blockIdx.x * blockDim.x + threadIdx.x;
    long total = 9L * O * Cin;
    if (i >= total) return;
    int ci = (int)(i % Cin);
    long rem = i / Cin;
    int o = (int)(rem % O);
    int tap = (int)(rem / O);
    wt[i] = (bf16_t)w[((long)o * Cin + ci) * 9 + tap];
}

__global__ void fill_zero_bf16(bf16_t* p, int n) {
    for (int i = threadIdx.x; i < n; i += blockDim.x) p[i] = (bf16_t)0.0f;
}

// ---------------- conv epilogue helper ----------------
__device__ __forceinline__
void conv_epilogue(const v8f& acc, const float* __restrict__ bias, int obase,
                   size_t oi, bf16_t* __restrict__ out_bf,
                   float* __restrict__ out_f, int relu_bf16out) {
    float vals[8];
#pragma unroll
    for (int r = 0; r < 8; ++r) vals[r] = acc[r] + bias[obase + r];
    if (relu_bf16out) {
        v8bf ov;
#pragma unroll
        for (int r = 0; r < 8; ++r) ov[r] = (bf16_t)fmaxf(vals[r], 0.0f);
        *(v8bf*)(out_bf + oi) = ov;
    } else {
        v4f o0 = {vals[0], vals[1], vals[2], vals[3]};
        v4f o1 = {vals[4], vals[5], vals[6], vals[7]};
        *(v4f*)(out_f + oi)     = o0;
        *(v4f*)(out_f + oi + 4) = o1;
    }
}

// ---------------- implicit-GEMM 3x3 SAME conv via WMMA bf16 ----------------
// in : [B][H][W][Cin]  bf16 (NHWC)
// wt : [9][O][Cin]     bf16
// out: [B][H][W][O]    bf16 (relu path) or f32 (final path)
// One wave computes a 32(out-ch) x 32(consecutive-W positions) block
// = 2x2 tiles of 16x16, reusing each A/B fragment twice (halves L2 traffic).
__global__ __launch_bounds__(256)
void conv3x3_wmma(const bf16_t* __restrict__ in,
                  const bf16_t* __restrict__ wt,
                  const float*  __restrict__ bias,
                  bf16_t* __restrict__ out_bf,
                  float*  __restrict__ out_f,
                  const bf16_t* __restrict__ zpad,
                  int H, int W, int Cin, int O, int B,
                  int relu_bf16out) {
    const int lane = threadIdx.x & 31;
    const int wave = threadIdx.x >> 5;
    const int gid  = blockIdx.x * (blockDim.x >> 5) + wave;

    const int posGroups = (H * W) >> 5;  // W multiple of 32 -> group stays in one row
    const int oGroups   = O >> 5;
    const int perBatch  = posGroups * oGroups;
    if (gid >= perBatch * B) return;     // wave-uniform

    const int b    = gid / perBatch;
    const int t    = gid % perBatch;
    const int og   = t / posGroups;
    const int pg   = t % posGroups;
    const int pos0 = pg << 5;
    const int row  = pos0 / W;
    const int col0 = pos0 % W;

    const int n     = lane & 15;   // N index (position) / A-matrix M row
    const int khalf = lane >> 4;   // K-half selector (contiguous k = 16*khalf + s)

    const bf16_t* inB = in + (size_t)b * H * W * Cin;
    const int kChunks = Cin >> 5;

    v8f acc00 = {}, acc01 = {}, acc10 = {}, acc11 = {};

    for (int kh = 0; kh < 3; ++kh) {
        const int irow  = row + kh - 1;
        const bool rowOK = (irow >= 0) && (irow < H);
        for (int kw = 0; kw < 3; ++kw) {
            const int ic0 = col0 + n + kw - 1;
            const int ic1 = ic0 + 16;
            const bool ok0 = rowOK && (ic0 >= 0) && (ic0 < W);
            const bool ok1 = rowOK && (ic1 >= 0) && (ic1 < W);
            // B fragments: lane holds N=n, contiguous K run [16*khalf, +16)
            const bf16_t* b0p = ok0
                ? (inB + ((size_t)irow * W + ic0) * Cin + khalf * 16) : zpad;
            const bf16_t* b1p = ok1
                ? (inB + ((size_t)irow * W + ic1) * Cin + khalf * 16) : zpad;
            // A fragments: lane holds M=n, two contiguous K runs of 8
            const bf16_t* a0p =
                wt + ((size_t)(kh * 3 + kw) * O + (og << 5) + n) * Cin + khalf * 16;
            const bf16_t* a1p = a0p + (size_t)16 * Cin;
            for (int kc = 0; kc < kChunks; ++kc) {
                v16bf b0 = *(const v16bf*)(b0p + kc * 32);
                v16bf b1 = *(const v16bf*)(b1p + kc * 32);
                v8bf a0lo = *(const v8bf*)(a0p + kc * 32);
                v8bf a0hi = *(const v8bf*)(a0p + kc * 32 + 8);
                v8bf a1lo = *(const v8bf*)(a1p + kc * 32);
                v8bf a1hi = *(const v8bf*)(a1p + kc * 32 + 8);
                v16bf a0 = __builtin_shufflevector(
                    a0lo, a0hi, 0, 1, 2, 3, 4, 5, 6, 7, 8, 9, 10, 11, 12, 13, 14, 15);
                v16bf a1 = __builtin_shufflevector(
                    a1lo, a1hi, 0, 1, 2, 3, 4, 5, 6, 7, 8, 9, 10, 11, 12, 13, 14, 15);
                acc00 = __builtin_amdgcn_wmma_f32_16x16x32_bf16(
                    false, a0, false, b0, (short)0, acc00, false, false);
                acc01 = __builtin_amdgcn_wmma_f32_16x16x32_bf16(
                    false, a0, false, b1, (short)0, acc01, false, false);
                acc10 = __builtin_amdgcn_wmma_f32_16x16x32_bf16(
                    false, a1, false, b0, (short)0, acc10, false, false);
                acc11 = __builtin_amdgcn_wmma_f32_16x16x32_bf16(
                    false, a1, false, b1, (short)0, acc11, false, false);
            }
        }
    }

    // Epilogue: lane holds position N=n, out channels M = r + 8*khalf (contiguous 8)
    const size_t posBase = (size_t)b * H * W + (size_t)row * W + col0 + n;
    const int ob0 = (og << 5) + khalf * 8;        // o-tile 0
    const int ob1 = ob0 + 16;                     // o-tile 1
    conv_epilogue(acc00, bias, ob0, posBase * O + ob0,
                  out_bf, out_f, relu_bf16out);
    conv_epilogue(acc01, bias, ob0, (posBase + 16) * O + ob0,
                  out_bf, out_f, relu_bf16out);
    conv_epilogue(acc10, bias, ob1, posBase * O + ob1,
                  out_bf, out_f, relu_bf16out);
    conv_epilogue(acc11, bias, ob1, (posBase + 16) * O + ob1,
                  out_bf, out_f, relu_bf16out);
}

// ---------------- gather + attention (tiny: ~0.4% of FLOPs) ----------------
// q  : [B][N][128] f32, k/v : [B][H*W][128] f32 (NHWC), u : [B][N] int
// One wave per (b,n); lane == h (H must be 32); C == 128.
__global__ __launch_bounds__(256)
void gather_attention(const float* __restrict__ q,
                      const float* __restrict__ k,
                      const float* __restrict__ v,
                      const int*   __restrict__ u,
                      float* __restrict__ out,
                      int B, int N, int H, int W, int C, float scale) {
    const int lane = threadIdx.x & 31;
    const int wid  = (int)(((long)blockIdx.x * blockDim.x + threadIdx.x) >> 5);
    if (wid >= B * N) return;   // wave-uniform
    const int b = wid / N;
    const int n = wid % N;

    int uu = u[(long)b * N + n];
    uu = uu < 0 ? 0 : (uu > W - 1 ? W - 1 : uu);

    const float* qp = q + ((long)b * N + n) * C;
    const float* kb = k + ((long)b * H * W + uu) * C;   // + h*W*C for row h
    const float* vb = v + ((long)b * H * W + uu) * C;

    // dot over C for this lane's h
    const float* kh = kb + (long)lane * W * C;
    float dot = 0.0f;
    for (int c = 0; c < C; c += 4) {
        v4f qv = *(const v4f*)(qp + c);
        v4f kv = *(const v4f*)(kh + c);
        dot += qv.x * kv.x + qv.y * kv.y + qv.z * kv.z + qv.w * kv.w;
    }
    dot *= scale;

    // softmax across the 32 lanes (= 32 heights)
    float m = dot;
#pragma unroll
    for (int off = 16; off > 0; off >>= 1) m = fmaxf(m, __shfl_xor(m, off, 32));
    float e = __expf(dot - m);
    float s = e;
#pragma unroll
    for (int off = 16; off > 0; off >>= 1) s += __shfl_xor(s, off, 32);
    const float wgt = e / s;

    // weighted sum over h; each lane owns 4 output channels
    v4f accv = {};
    for (int h = 0; h < 32; ++h) {
        float wh = __shfl(wgt, h, 32);
        v4f vv = *(const v4f*)(vb + (long)h * W * C + lane * 4);
        accv.x += wh * vv.x;
        accv.y += wh * vv.y;
        accv.z += wh * vv.z;
        accv.w += wh * vv.w;
    }
    *(v4f*)(out + ((long)b * N + n) * C + lane * 4) = accv;
}

// ---------------- host launcher ----------------
extern "C" void kernel_launch(void* const* d_in, const int* in_sizes, int n_in,
                              void* d_out, int out_size, void* d_ws, size_t ws_size,
                              hipStream_t stream) {
    (void)in_sizes; (void)n_in; (void)out_size; (void)ws_size;

    const float* x   = (const float*)d_in[0];
    const float* y   = (const float*)d_in[1];
    const float* z   = (const float*)d_in[2];
    const int*   u   = (const int*)  d_in[3];
    const float* wq1 = (const float*)d_in[4];
    const float* bq1 = (const float*)d_in[5];
    const float* wq2 = (const float*)d_in[6];
    const float* bq2 = (const float*)d_in[7];
    const float* wk1 = (const float*)d_in[8];
    const float* bk1 = (const float*)d_in[9];
    const float* wk2 = (const float*)d_in[10];
    const float* bk2 = (const float*)d_in[11];
    float* out = (float*)d_out;

    const int B = 2, C = 256, S = 64, H = 32, W = 512, MID = 128, CZ = 128;
    const int NQ = S * S;     // 4096
    const int NK = H * W;     // 16384

    // workspace carve-out
    char* ws = (char*)d_ws;
    size_t off = 0;
    auto take = [&](size_t bytes) -> char* {
        char* p = ws + off;
        off = (off + bytes + 255) & ~(size_t)255;
        return p;
    };
    bf16_t* xb   = (bf16_t*)take((size_t)B * NQ * C   * 2);
    bf16_t* yb   = (bf16_t*)take((size_t)B * NK * C   * 2);
    float*  zt   = (float*) take((size_t)B * NK * CZ  * 4);
    bf16_t* wq1t = (bf16_t*)take((size_t)9 * MID * C   * 2);
    bf16_t* wq2t = (bf16_t*)take((size_t)9 * MID * MID * 2);
    bf16_t* wk1t = (bf16_t*)take((size_t)9 * MID * C   * 2);
    bf16_t* wk2t = (bf16_t*)take((size_t)9 * MID * MID * 2);
    bf16_t* qmid = (bf16_t*)take((size_t)B * NQ * MID * 2);
    bf16_t* kmid = (bf16_t*)take((size_t)B * NK * MID * 2);
    float*  qf   = (float*) take((size_t)B * NQ * MID * 4);
    float*  kf   = (float*) take((size_t)B * NK * MID * 4);
    bf16_t* zpad = (bf16_t*)take(2048);

    const dim3 blk(256);
    auto grid1 = [](long n) { return dim3((unsigned)((n + 255) / 256)); };

    // layout transforms
    nchw_to_nhwc_bf16<<<grid1((long)B * C  * NQ), blk, 0, stream>>>(x, xb, B, C, NQ);
    nchw_to_nhwc_bf16<<<grid1((long)B * C  * NK), blk, 0, stream>>>(y, yb, B, C, NK);
    nchw_to_nhwc_f32 <<<grid1((long)B * CZ * NK), blk, 0, stream>>>(z, zt, B, CZ, NK);
    weight_transform<<<grid1(9L * MID * C  ), blk, 0, stream>>>(wq1, wq1t, MID, C);
    weight_transform<<<grid1(9L * MID * MID), blk, 0, stream>>>(wq2, wq2t, MID, MID);
    weight_transform<<<grid1(9L * MID * C  ), blk, 0, stream>>>(wk1, wk1t, MID, C);
    weight_transform<<<grid1(9L * MID * MID), blk, 0, stream>>>(wk2, wk2t, MID, MID);
    fill_zero_bf16<<<1, 256, 0, stream>>>(zpad, 1024);

    // conv chains: one wave per 32x32 block (8 waves/block; counts divide exactly)
    const int wavesQ = B * (MID / 32) * (NQ / 32);   // 1024
    const int wavesK = B * (MID / 32) * (NK / 32);   // 4096
    conv3x3_wmma<<<dim3(wavesQ / 8), blk, 0, stream>>>(
        xb, wq1t, bq1, qmid, nullptr, zpad, S, S, C, MID, B, 1);
    conv3x3_wmma<<<dim3(wavesQ / 8), blk, 0, stream>>>(
        qmid, wq2t, bq2, nullptr, qf, zpad, S, S, MID, MID, B, 0);
    conv3x3_wmma<<<dim3(wavesK / 8), blk, 0, stream>>>(
        yb, wk1t, bk1, kmid, nullptr, zpad, H, W, C, MID, B, 1);
    conv3x3_wmma<<<dim3(wavesK / 8), blk, 0, stream>>>(
        kmid, wk2t, bk2, nullptr, kf, zpad, H, W, MID, MID, B, 0);

    // gather + attention
    const float scale = 1.0f / sqrtf((float)MID);
    gather_attention<<<dim3((B * NQ) / 8), blk, 0, stream>>>(
        qf, kf, zt, u, out, B, NQ, H, W, CZ, scale);
}
